// AttentionHead_33311766348286
// MI455X (gfx1250) — matrix operand order repaired and checked by reading, not compiled, not run on hardware
//
#include <hip/hip_runtime.h>
#include <hip/hip_bf16.h>

#define SEQ 8192
#define DM  1024
#define HD  64
#define NS  8        // query-axis splits for the stats pass
#define NJ  8        // key-axis splits for the output pass

typedef __bf16 bf16_t;
typedef __attribute__((ext_vector_type(16))) __bf16 v16bf;
typedef __attribute__((ext_vector_type(8)))  __bf16 v8bf;
typedef __attribute__((ext_vector_type(2)))  __bf16 bf16x2;
typedef __attribute__((ext_vector_type(8)))  float  v8f;

static __device__ inline v8f zero8() {
    v8f z = {0.f, 0.f, 0.f, 0.f, 0.f, 0.f, 0.f, 0.f};
    return z;
}

static __device__ inline v8f wmma_bf16(v16bf a, v16bf b, v8f c) {
    return __builtin_amdgcn_wmma_f32_16x16x32_bf16(
        /*neg_a=*/false, a, /*neg_b=*/false, b,
        /*c_mod=*/(short)0, c, /*reuse_a=*/false, /*reuse_b=*/false);
}

// ---- Fragment loaders (contiguous k order: k = 16*(lane/16) + element, same for A and B) ----

// A fragment 16x32 (MxK) bf16 from row-major bf16 [rows x ld].
static __device__ inline v16bf load_a_bf16(const bf16_t* base, int ld, int row0, int k0) {
    int lane = threadIdx.x & 31;
    int m    = lane & 15;
    int kb   = (lane >> 4) << 4;          // 0 or 16
    const bf16_t* p = base + (row0 + m) * ld + k0 + kb;
    v16bf f;
#pragma unroll
    for (int r = 0; r < 8; ++r) {
        int kk = 2 * r;   // 0,2,...,14
        bf16x2 t = *(const bf16x2*)(p + kk);
        f[2 * r]     = t.x;
        f[2 * r + 1] = t.y;
    }
    return f;
}

// Same A layout from fp32 row-major, converting on the fly.
static __device__ inline v16bf load_a_f32(const float* base, int ld, int row0, int k0) {
    int lane = threadIdx.x & 31;
    int m    = lane & 15;
    int kb   = (lane >> 4) << 4;
    const float* p = base + (row0 + m) * ld + k0 + kb;
    v16bf f;
#pragma unroll
    for (int r = 0; r < 8; ++r) {
        int kk = 2 * r;
        float2 t = *(const float2*)(p + kk);
        f[2 * r]     = (bf16_t)t.x;
        f[2 * r + 1] = (bf16_t)t.y;
    }
    return f;
}

// B fragment 32x16 (KxN) where B = src^T : B(kk,n) = src[(n0+n)*ld + k0+kk].
// Contiguous along K -> bf16x2 chunk loads.
static __device__ inline v16bf load_bT_bf16(const bf16_t* src, int ld, int n0, int k0) {
    int lane = threadIdx.x & 31;
    int n    = lane & 15;
    int kb   = (lane >> 4) << 4;          // 0 or 16
    const bf16_t* p = src + (size_t)(n0 + n) * ld + k0 + kb;
    v16bf f;
#pragma unroll
    for (int r = 0; r < 8; ++r) {
        bf16x2 t = *(const bf16x2*)(p + 2 * r);
        f[2 * r]     = t.x;
        f[2 * r + 1] = t.y;
    }
    return f;
}

// B fragment 32x16 from row-major fp32 [K x N]: B(kk,n)=src[(k0+kk)*ld + n0+n]
static __device__ inline v16bf load_b_f32(const float* src, int ld, int k0, int n0) {
    int lane = threadIdx.x & 31;
    int n    = (lane & 15) + n0;
    int kb   = ((lane >> 4) << 4) + k0;
    v16bf f;
#pragma unroll
    for (int r = 0; r < 8; ++r) {
        f[2 * r]     = (bf16_t)src[(kb + 2 * r) * ld + n];
        f[2 * r + 1] = (bf16_t)src[(kb + 2 * r + 1) * ld + n];
    }
    return f;
}

// ---- Kernel 1: projections. Q,K stored row-major bf16; V stored TRANSPOSED
// (vbT[HD][SEQ]) so the p@v pass gets contiguous-K B fragments and this kernel
// gets a single contiguous b128 store per lane per tile. ----
__global__ __launch_bounds__(128) void proj_kernel(
    const float* __restrict__ Q, const float* __restrict__ K,
    const float* __restrict__ V, const float* __restrict__ Wq,
    const float* __restrict__ Wk, const float* __restrict__ Wv,
    bf16_t* __restrict__ Pq, bf16_t* __restrict__ Pk, bf16_t* __restrict__ PvT) {
    const float* X;
    const float* W;
    bf16_t* P;
    bool trans;
    if (blockIdx.y == 0)      { X = Q; W = Wq; P = Pq;  trans = false; }
    else if (blockIdx.y == 1) { X = K; W = Wk; P = Pk;  trans = false; }
    else                      { X = V; W = Wv; P = PvT; trans = true;  }

    int wave = threadIdx.x >> 5;
    int lane = threadIdx.x & 31;
    int row0 = blockIdx.x * 64 + wave * 16;

    v8f acc[4];
#pragma unroll
    for (int t = 0; t < 4; ++t) acc[t] = zero8();

    for (int k0 = 0; k0 < DM; k0 += 32) {
        v16bf a = load_a_f32(X, DM, row0, k0);
#pragma unroll
        for (int t = 0; t < 4; ++t) {
            v16bf b = load_b_f32(W, HD, k0, t * 16);
            acc[t]  = wmma_bf16(a, b, acc[t]);
        }
    }

    int n  = lane & 15;
    int mh = (lane >> 4) * 8;
    if (!trans) {
#pragma unroll
        for (int t = 0; t < 4; ++t)
#pragma unroll
            for (int r = 0; r < 8; ++r)
                P[(row0 + mh + r) * HD + t * 16 + n] = (bf16_t)acc[t][r];
    } else {
#pragma unroll
        for (int t = 0; t < 4; ++t) {
            v8bf pk;
#pragma unroll
            for (int r = 0; r < 8; ++r) pk[r] = (bf16_t)acc[t][r];
            *(v8bf*)(P + (size_t)(t * 16 + n) * SEQ + row0 + mh) = pk;  // 16B store
        }
    }
}

// ---- Kernel 2: partial column softmax stats (query axis split NS ways).
// Each wave: 32 key columns (4 persistent B frags) x SEQ/NS query rows. ----
__global__ __launch_bounds__(128) void stats_kernel(
    const bf16_t* __restrict__ qb, const bf16_t* __restrict__ kb,
    float* __restrict__ pmax, float* __restrict__ psum) {
    int wave = threadIdx.x >> 5;
    int lane = threadIdx.x & 31;
    int j0   = (blockIdx.x * 4 + wave) * 32;
    int s    = blockIdx.y;
    int ibeg = s * (SEQ / NS);
    int iend = ibeg + (SEQ / NS);

    v16bf b00 = load_bT_bf16(kb, HD, j0, 0);
    v16bf b01 = load_bT_bf16(kb, HD, j0, 32);
    v16bf b10 = load_bT_bf16(kb, HD, j0 + 16, 0);
    v16bf b11 = load_bT_bf16(kb, HD, j0 + 16, 32);

    float m0 = -1e30f, l0 = 0.f, m1 = -1e30f, l1 = 0.f;
    for (int i0 = ibeg; i0 < iend; i0 += 16) {
        __builtin_prefetch(qb + (i0 + 32) * HD, 0, 1);
        v16bf a0 = load_a_bf16(qb, HD, i0, 0);
        v16bf a1 = load_a_bf16(qb, HD, i0, 32);
#pragma unroll
        for (int h = 0; h < 2; ++h) {
            v8f c = zero8();
            c = wmma_bf16(a0, h ? b10 : b00, c);
            c = wmma_bf16(a1, h ? b11 : b01, c);
            float m = h ? m1 : m0, l = h ? l1 : l0;
            float tm = -1e30f;
#pragma unroll
            for (int r = 0; r < 8; ++r) {
                float v = c[r] * 0.125f;   // /sqrt(64)
                c[r]    = v;
                tm      = fmaxf(tm, v);
            }
            float mn  = fmaxf(m, tm);
            float acc = l * __expf(m - mn);
#pragma unroll
            for (int r = 0; r < 8; ++r) acc += __expf(c[r] - mn);
            if (h) { m1 = mn; l1 = acc; } else { m0 = mn; l0 = acc; }
        }
    }
#pragma unroll
    for (int h = 0; h < 2; ++h) {
        float m = h ? m1 : m0, l = h ? l1 : l0;
        float mo = __shfl_xor(m, 16, 32);     // other 8-row half of same column
        float lo = __shfl_xor(l, 16, 32);
        float mm = fmaxf(m, mo);
        float ll = l * __expf(m - mm) + lo * __expf(mo - mm);
        if (lane < 16) {
            pmax[(size_t)s * SEQ + j0 + h * 16 + lane] = mm;
            psum[(size_t)s * SEQ + j0 + h * 16 + lane] = ll;
        }
    }
}

// ---- Kernel 2b: merge NS partial (m,l) per column ----
__global__ __launch_bounds__(256) void stats_reduce(
    const float* __restrict__ pmax, const float* __restrict__ psum,
    float* __restrict__ cmax, float* __restrict__ csum) {
    int j = blockIdx.x * 256 + threadIdx.x;
    float mm = -1e30f;
#pragma unroll
    for (int s = 0; s < NS; ++s) mm = fmaxf(mm, pmax[(size_t)s * SEQ + j]);
    float ll = 0.f;
#pragma unroll
    for (int s = 0; s < NS; ++s)
        ll += psum[(size_t)s * SEQ + j] * __expf(pmax[(size_t)s * SEQ + j] - mm);
    cmax[j] = mm;
    csum[j] = ll;
}

// ---- Kernel 3: partial out = softmax_cols(qk^T/8) @ v, key axis split NJ ways ----
__global__ __launch_bounds__(128) void out_kernel(
    const bf16_t* __restrict__ qb, const bf16_t* __restrict__ kb,
    const bf16_t* __restrict__ vbT, const float* __restrict__ colmax,
    const float* __restrict__ colsum, float* __restrict__ pout) {
    __shared__ bf16_t ptile[4][16 * 32];   // per-wave 16(i) x 32(j) prob tile

    int wave = threadIdx.x >> 5;
    int lane = threadIdx.x & 31;
    int i0   = (blockIdx.x * 4 + wave) * 16;
    int js   = blockIdx.y;
    int jbeg = js * (SEQ / NJ);
    int jend = jbeg + (SEQ / NJ);
    int n    = lane & 15;
    int mh   = (lane >> 4) * 8;

    v16bf qa0 = load_a_bf16(qb, HD, i0, 0);    // persistent q fragments
    v16bf qa1 = load_a_bf16(qb, HD, i0, 32);

    v8f acc[4];
#pragma unroll
    for (int t = 0; t < 4; ++t) acc[t] = zero8();

    bf16_t* myp = &ptile[wave][0];

    for (int j0 = jbeg; j0 < jend; j0 += 32) {
        __builtin_prefetch(kb + (j0 + 32) * HD, 0, 1);
#pragma unroll
        for (int h = 0; h < 2; ++h) {
            int jn = j0 + h * 16;
            v16bf kb0 = load_bT_bf16(kb, HD, jn, 0);
            v16bf kb1 = load_bT_bf16(kb, HD, jn, 32);
            v8f sc = zero8();
            sc = wmma_bf16(qa0, kb0, sc);
            sc = wmma_bf16(qa1, kb1, sc);
            float mj = colmax[jn + n];
            float rl = 1.0f / colsum[jn + n];
#pragma unroll
            for (int r = 0; r < 8; ++r) {
                float p = __expf(sc[r] * 0.125f - mj) * rl;
                myp[(mh + r) * 32 + h * 16 + n] = (bf16_t)p;
            }
        }
        __syncthreads();   // C-layout -> A-layout via LDS round trip

        v16bf pa = load_a_bf16(myp, 32, 0, 0);
#pragma unroll
        for (int t = 0; t < 4; ++t) {
            // contiguous-K B fragment thanks to transposed V
            v16bf bv = load_bT_bf16(vbT, SEQ, t * 16, j0);
            acc[t]   = wmma_bf16(pa, bv, acc[t]);
        }
        __syncthreads();
    }

    float* po = pout + (size_t)js * SEQ * HD;
#pragma unroll
    for (int t = 0; t < 4; ++t)
#pragma unroll
        for (int r = 0; r < 8; ++r)
            po[(i0 + mh + r) * HD + t * 16 + n] = acc[t][r];
}

// ---- Kernel 3b: sum NJ output partials (deterministic tree-free sum) ----
__global__ __launch_bounds__(256) void out_reduce(
    const float* __restrict__ pout, float* __restrict__ out) {
    int idx = blockIdx.x * 256 + threadIdx.x;      // over SEQ*HD/4 float4 slots
    const size_t stride4 = (size_t)SEQ * HD / 4;
    float4 s;
    s.x = 0.f; s.y = 0.f; s.z = 0.f; s.w = 0.f;
#pragma unroll
    for (int js = 0; js < NJ; ++js) {
        float4 p = ((const float4*)pout)[(size_t)js * stride4 + idx];
        s.x += p.x; s.y += p.y; s.z += p.z; s.w += p.w;
    }
    ((float4*)out)[idx] = s;
}

extern "C" void kernel_launch(void* const* d_in, const int* in_sizes, int n_in,
                              void* d_out, int out_size, void* d_ws, size_t ws_size,
                              hipStream_t stream) {
    (void)in_sizes; (void)n_in; (void)out_size; (void)ws_size;
    const float* Q  = (const float*)d_in[0];
    const float* K  = (const float*)d_in[1];
    const float* V  = (const float*)d_in[2];
    const float* Wq = (const float*)d_in[3];
    const float* Wk = (const float*)d_in[4];
    const float* Wv = (const float*)d_in[5];

    // Workspace layout (~19.5 MB, L2-resident)
    bf16_t* qb  = (bf16_t*)d_ws;                       // [SEQ][HD]
    bf16_t* kb  = qb + (size_t)SEQ * HD;               // [SEQ][HD]
    bf16_t* vbT = kb + (size_t)SEQ * HD;               // [HD][SEQ] (transposed)
    float* pmax = (float*)(vbT + (size_t)SEQ * HD);    // [NS][SEQ]
    float* psum = pmax + (size_t)NS * SEQ;             // [NS][SEQ]
    float* cmax = psum + (size_t)NS * SEQ;             // [SEQ]
    float* csum = cmax + SEQ;                          // [SEQ]
    float* pout = csum + SEQ;                          // [NJ][SEQ][HD]

    proj_kernel<<<dim3(SEQ / 64, 3), 128, 0, stream>>>(Q, K, V, Wq, Wk, Wv, qb, kb, vbT);
    stats_kernel<<<dim3(SEQ / 128, NS), 128, 0, stream>>>(qb, kb, pmax, psum);
    stats_reduce<<<dim3(SEQ / 256), 256, 0, stream>>>(pmax, psum, cmax, csum);
    out_kernel<<<dim3(SEQ / 64, NJ), 128, 0, stream>>>(qb, kb, vbT, cmax, csum, pout);
    out_reduce<<<dim3((SEQ * HD / 4) / 256), 256, 0, stream>>>(pout, (float*)d_out);
}
